// AttentionLayer_31164282699946
// MI455X (gfx1250) — matrix operand order repaired and checked by reading, not compiled, not run on hardware
//
#include <hip/hip_runtime.h>
#include <hip/hip_bf16.h>
#include <stdint.h>

// Problem constants (match reference).
static constexpr int BATCH = 4;
static constexpr int CIN   = 64;
static constexpr int COUTC = 64;
static constexpr int DKC   = 64;
static constexpr int NSEQ  = 4096;

typedef __attribute__((ext_vector_type(16))) __bf16 v16bf;
typedef __attribute__((ext_vector_type(8)))  __bf16 v8bf;
typedef __attribute__((ext_vector_type(8)))  float  v8f;
typedef __attribute__((ext_vector_type(4)))  int    v4i;

static __device__ __forceinline__ v8f wmma_bf16(v16bf a, v16bf b, v8f c) {
  // D = A(16x32 bf16) x B(32x16 bf16) + C(16x16 f32)
  return __builtin_amdgcn_wmma_f32_16x16x32_bf16(false, a, false, b,
                                                 (short)0, c, false, false);
}

// 16-byte global -> LDS copy; async (ASYNCcnt) path when the builtin exists.
// Builtin param types (from hipcc diagnostic): pointer-to-v4i, global/LDS.
static __device__ __forceinline__ void async_copy16(const __bf16* g, __bf16* l) {
#if __has_builtin(__builtin_amdgcn_global_load_async_to_lds_b128)
  __builtin_amdgcn_global_load_async_to_lds_b128(
      (__attribute__((address_space(1))) v4i*)(void*)(g),
      (__attribute__((address_space(3))) v4i*)(void*)(l), 0, 0);
#else
  *(uint4*)l = *(const uint4*)g;
#endif
}

static __device__ __forceinline__ void wait_async0() {
#if __has_builtin(__builtin_amdgcn_s_wait_asynccnt)
  __builtin_amdgcn_s_wait_asynccnt(0);
#else
  asm volatile("s_wait_asynccnt 0x0" ::: "memory");
#endif
}

// Build a 16x32 bf16 A-operand for one lane from an LDS row pointer.
// Per ISA 7.12.2 (16-bit A 16x32): lanes 0-15 row=lane, K in {kb8..kb8+7} U {16+kb8..16+kb8+7}
// with kb8 = (lane>=16)?8:0. Both 8-element groups are contiguous -> two b128 LDS loads.
static __device__ __forceinline__ v16bf load_a(const __bf16* base, int kb8) {
  v8bf lo = *(const v8bf*)(base + kb8);
  v8bf hi = *(const v8bf*)(base + 16 + kb8);
  v16bf r;
#pragma unroll
  for (int j = 0; j < 8; ++j) { r[j] = lo[j]; r[8 + j] = hi[j]; }
  return r;
}

// ---------------------------------------------------------------------------
// Kernel 1: 1x1-conv projections, f32 in -> bf16 out, WMMA-friendly layouts.
//   Qb[b][n][d] (row-major n,d)   -- A operand rows = n
//   Kb[b][d][m]                   -- B operand rows = d
//   Vb[b][o][n]                   -- A operand rows = o, K = n
// ---------------------------------------------------------------------------
__global__ __launch_bounds__(256) void proj_kernel(
    const float* __restrict__ x,  const float* __restrict__ Wq,
    const float* __restrict__ Wk, const float* __restrict__ Wv,
    __bf16* __restrict__ Qb, __bf16* __restrict__ Kb, __bf16* __restrict__ Vb)
{
  __shared__ float xs[CIN * 64];                     // 16 KB x-tile
  __shared__ float wq[DKC * CIN];                    // 16 KB
  __shared__ float wk[DKC * CIN];                    // 16 KB
  __shared__ float wv[COUTC * CIN];                  // 16 KB  (total 64 KB)

  const int b  = blockIdx.y;
  const int n0 = blockIdx.x * 64;
  const int t  = threadIdx.x;

#pragma unroll
  for (int i = 0; i < 16; ++i) {
    const int idx = t + i * 256;                     // idx = c*64 + nl
    const int c  = idx >> 6;
    const int nl = idx & 63;
    xs[idx] = x[(size_t)b * CIN * NSEQ + (size_t)c * NSEQ + n0 + nl];
    wq[idx] = Wq[idx];
    wk[idx] = Wk[idx];
    wv[idx] = Wv[idx];
  }
  __syncthreads();

  const int nl = t & 63;
  const int d0 = t >> 6;                             // 0..3
  for (int d = d0; d < DKC; d += 4) {
    float aq = 0.f, ak = 0.f, av = 0.f;
#pragma unroll
    for (int c = 0; c < CIN; ++c) {
      const float xv = xs[c * 64 + nl];
      aq = fmaf(wq[d * CIN + c], xv, aq);
      ak = fmaf(wk[d * CIN + c], xv, ak);
      av = fmaf(wv[d * CIN + c], xv, av);
    }
    const int n = n0 + nl;
    Qb[(size_t)b * NSEQ * DKC  + (size_t)n * DKC  + d] = (__bf16)aq;
    Kb[(size_t)b * DKC  * NSEQ + (size_t)d * NSEQ + n] = (__bf16)ak;
    Vb[(size_t)b * COUTC* NSEQ + (size_t)d * NSEQ + n] = (__bf16)av;
  }
}

// ---------------------------------------------------------------------------
// Kernel 2: fused flash attention (softmax over n, per output column m).
// Block = 4 waves; wave w owns columns m0 = blockIdx.x*64 + w*16.
// Streams n in chunks of 32; scores never hit memory.
// ---------------------------------------------------------------------------
__global__ __launch_bounds__(128) void flash_attn_kernel(
    const __bf16* __restrict__ Qb, const __bf16* __restrict__ Kb,
    const __bf16* __restrict__ Vb, float* __restrict__ out)
{
  __shared__ __align__(16) __bf16 q_lds[32 * DKC];     // 4 KB  (shared by 4 waves)
  __shared__ __align__(16) __bf16 v_lds[COUTC * 32];   // 4 KB
  __shared__ __align__(16) __bf16 p_lds[4][16 * 32];   // 4 KB  (per-wave pad, [m][k])

  const int b    = blockIdx.y;
  const int tid  = threadIdx.x;
  const int wv   = tid >> 5;
  const int lane = tid & 31;
  const int hf   = lane >> 4;        // half-wave id (0/1)
  const int l16  = lane & 15;
  const int m0   = blockIdx.x * 64 + wv * 16;

  // Persistent B operand: K[d, m0+l16] for two 32-deep d-chunks.
  // B 32x16 layout: col=lane%16; lanes 0-15 hold K=0..7,16..23; lanes 16-31 hold K=8..15,24..31.
  v16bf bk0, bk1;
  {
    const __bf16* kb = Kb + (size_t)b * DKC * NSEQ + m0 + l16;
    const int dOff = hf * 8;
#pragma unroll
    for (int j = 0; j < 16; ++j) {
      bk0[j] = kb[(size_t)(dOff + j + (j >= 8 ? 8 : 0)) * NSEQ];
      bk1[j] = kb[(size_t)(32 + dOff + j + (j >= 8 ? 8 : 0)) * NSEQ];
    }
  }

  v8f o0 = {}, o1 = {}, o2 = {}, o3 = {};
  float m_run = -1e30f, l_run = 0.0f;

  const __bf16* qbase = Qb + (size_t)b * NSEQ * DKC;
  const __bf16* vbase = Vb + (size_t)b * COUTC * NSEQ;

  for (int n0 = 0; n0 < NSEQ; n0 += 32) {
    // -- cooperative async tile loads --------------------------------------
    {
      const __bf16* g = qbase + n0 * DKC;            // 2048 contiguous bf16
      async_copy16(g + tid * 8,        &q_lds[tid * 8]);
      async_copy16(g + 1024 + tid * 8, &q_lds[1024 + tid * 8]);
#pragma unroll
      for (int c = tid; c < 256; c += 128) {         // V rows: 64 x 32 bf16
        const int row = c >> 2, part = c & 3;
        async_copy16(vbase + (size_t)row * NSEQ + n0 + part * 8,
                     &v_lds[row * 32 + part * 8]);
      }
      if (n0 + 32 < NSEQ) __builtin_prefetch(qbase + (n0 + 32) * DKC, 0, 1);
    }
    wait_async0();
    __syncthreads();

    // -- per-wave A operands (ds_load_b128 pairs) --------------------------
    const int kb8 = hf * 8;
    const __bf16* r0 = &q_lds[l16 * DKC];
    const __bf16* r1 = &q_lds[(16 + l16) * DKC];
    v16bf aq00 = load_a(r0,      kb8);
    v16bf aq01 = load_a(r0 + 32, kb8);
    v16bf aq10 = load_a(r1,      kb8);
    v16bf aq11 = load_a(r1 + 32, kb8);
    v16bf avA  = load_a(&v_lds[( 0 + l16) * 32], kb8);
    v16bf avB  = load_a(&v_lds[(16 + l16) * 32], kb8);
    v16bf avC  = load_a(&v_lds[(32 + l16) * 32], kb8);
    v16bf avD  = load_a(&v_lds[(48 + l16) * 32], kb8);
    __syncthreads();   // all waves done reading q/v before next overwrite

    // -- scores: S[n,m] = Q(n,:) . K(:,m) ----------------------------------
    v8f zero = {};
    v8f s0 = wmma_bf16(aq00, bk0, zero);
    s0     = wmma_bf16(aq01, bk1, s0);
    v8f s1 = wmma_bf16(aq10, bk0, zero);
    s1     = wmma_bf16(aq11, bk1, s1);

    // -- online softmax over n (rows). Each lane owns column m0+l16; rows
    //    hf*8+j (s0) and 16+hf*8+j (s1). Cross-half via shfl_xor(16).
    float t0[8], t1[8];
    float tmax = -1e30f;
#pragma unroll
    for (int j = 0; j < 8; ++j) {
      t0[j] = s0[j] * 0.125f;                        // 1/sqrt(dk)
      t1[j] = s1[j] * 0.125f;
      tmax  = fmaxf(tmax, fmaxf(t0[j], t1[j]));
    }
    tmax = fmaxf(tmax, __shfl_xor(tmax, 16, 32));
    const float mnew  = fmaxf(m_run, tmax);
    const float scale = __expf(m_run - mnew);
    float psum = 0.f;
    v8bf p0, p1;
#pragma unroll
    for (int j = 0; j < 8; ++j) {
      const float e0 = __expf(t0[j] - mnew);
      const float e1 = __expf(t1[j] - mnew);
      psum += e0 + e1;
      p0[j] = (__bf16)e0;
      p1[j] = (__bf16)e1;
    }
    psum += __shfl_xor(psum, 16, 32);
    l_run = l_run * scale + psum;
    m_run = mnew;
    o0 *= scale; o1 *= scale; o2 *= scale; o3 *= scale;

    // -- bounce P through per-wave LDS pad to form the B operand -----------
    {
      __bf16* prow = &p_lds[wv][l16 * 32];           // [m][k], k = n_local
      *(v8bf*)(prow + hf * 8)      = p0;             // rows hf*8..hf*8+7
      *(v8bf*)(prow + 16 + hf * 8) = p1;             // rows 16+hf*8..+7
    }
    asm volatile("s_wait_dscnt 0x0" ::: "memory");   // wave-internal xlane dep
    v16bf bp;
    {
      const __bf16* pr = &p_lds[wv][l16 * 32 + hf * 8];
      v8bf lo = *(const v8bf*)(pr);
      v8bf hi = *(const v8bf*)(pr + 16);
#pragma unroll
      for (int j = 0; j < 8; ++j) { bp[j] = lo[j]; bp[8 + j] = hi[j]; }
    }

    // -- O += V(o, n-chunk) x P(n-chunk, m) --------------------------------
    o0 = wmma_bf16(avA, bp, o0);
    o1 = wmma_bf16(avB, bp, o1);
    o2 = wmma_bf16(avC, bp, o2);
    o3 = wmma_bf16(avD, bp, o3);
  }

  // -- epilogue: out[b, o, m] = O / l ---------------------------------------
  const float inv_l = 1.0f / l_run;
  float* ob = out + (size_t)b * COUTC * NSEQ + (m0 + l16);
#pragma unroll
  for (int j = 0; j < 8; ++j) {
    ob[( 0 + hf * 8 + j) * NSEQ] = o0[j] * inv_l;
    ob[(16 + hf * 8 + j) * NSEQ] = o1[j] * inv_l;
    ob[(32 + hf * 8 + j) * NSEQ] = o2[j] * inv_l;
    ob[(48 + hf * 8 + j) * NSEQ] = o3[j] * inv_l;
  }
}

// ---------------------------------------------------------------------------
extern "C" void kernel_launch(void* const* d_in, const int* in_sizes, int n_in,
                              void* d_out, int out_size, void* d_ws, size_t ws_size,
                              hipStream_t stream) {
  const float* x  = (const float*)d_in[0];
  const float* Wq = (const float*)d_in[1];
  const float* Wk = (const float*)d_in[2];
  const float* Wv = (const float*)d_in[3];

  // bf16 staging in workspace: 3 * 4*4096*64 * 2B = 6 MB
  __bf16* Qb = (__bf16*)d_ws;
  __bf16* Kb = Qb + (size_t)BATCH * NSEQ * DKC;
  __bf16* Vb = Kb + (size_t)BATCH * DKC * NSEQ;

  proj_kernel<<<dim3(NSEQ / 64, BATCH), 256, 0, stream>>>(x, Wq, Wk, Wv, Qb, Kb, Vb);
  flash_attn_kernel<<<dim3(NSEQ / 64, BATCH), 128, 0, stream>>>(Qb, Kb, Vb, (float*)d_out);
}
